// MultiheadAttention_28509992911313
// MI455X (gfx1250) — matrix operand order repaired and checked by reading, not compiled, not run on hardware
//
#include <hip/hip_runtime.h>

typedef __bf16 bf16_t;
typedef __attribute__((ext_vector_type(16))) __bf16 v16bf;
typedef __attribute__((ext_vector_type(8)))  float  v8f;

union FragBF {
    v16bf v;
    uint4 q[2];
};

__device__ __forceinline__ v8f wmma_bf16(v16bf a, v16bf b, v8f c) {
    // D = A(16x32 bf16) x B(32x16 bf16) + C(16x16 f32)
    return __builtin_amdgcn_wmma_f32_16x16x32_bf16(
        /*neg_a=*/false, a, /*neg_b=*/false, b,
        /*c_mod=*/(short)0, c, /*reuse_a=*/false, /*reuse_b=*/false);
}

// CDNA5 async global->LDS copy (ASYNCcnt-tracked, no VGPR data movement).
// GV mode: 64-bit VGPR address, LDS byte offset in a VGPR.
__device__ __forceinline__ void async_g2l_b128(const void* gptr, void* lptr) {
    asm volatile("global_load_async_to_lds_b128 %0, %1, off"
                 :
                 : "v"((unsigned)(unsigned long long)lptr), "v"(gptr)
                 : "memory");
}
#define WAIT_ASYNC(n) asm volatile("s_wait_asynccnt %0" ::"n"(n) : "memory")

// One-step XOR-butterfly exchange via ds_swizzle (group-of-32 mode).
#define SWZ_F(v, pat) \
    __int_as_float(__builtin_amdgcn_ds_swizzle(__float_as_int(v), (pat)))
#define SWZ_X1 0x041F
#define SWZ_X2 0x081F
#define SWZ_X4 0x101F
#define SWZ_X8 0x201F

// ---------------------------------------------------------------------------
// Split fp32 -> bf16 hi + bf16 lo (error-compensated pair for bf16x3 GEMM)
// ---------------------------------------------------------------------------
__global__ void split_f32(const float* __restrict__ in,
                          bf16_t* __restrict__ hi, bf16_t* __restrict__ lo,
                          int n) {
    for (int i = blockIdx.x * blockDim.x + threadIdx.x; i < n;
         i += gridDim.x * blockDim.x) {
        float v = in[i];
        bf16_t h = (bf16_t)v;
        hi[i] = h;
        lo[i] = (bf16_t)(v - (float)h);
    }
}

// ---------------------------------------------------------------------------
// C[M,N] = A[M,K] @ W[N,K]^T + bias   via bf16x3 (hi*hi + hi*lo + lo*hi)
// Block tile 64x64, BK=32, 256 threads (8 waves), wave tile 16x32.
// Tiles staged with async global->LDS loads, double-buffered.
// ---------------------------------------------------------------------------
#define LDT 40  // padded LDS row stride (elements) for 32-wide k tiles

template <typename OUT>
__global__ __launch_bounds__(256) void gemm_nt_bf16x3(
    const bf16_t* __restrict__ Ahi, const bf16_t* __restrict__ Alo,
    const bf16_t* __restrict__ Whi, const bf16_t* __restrict__ Wlo,
    const float* __restrict__ bias, OUT* __restrict__ C,
    int M, int N, int K) {
    __shared__ bf16_t sAh[2][64 * LDT];
    __shared__ bf16_t sAl[2][64 * LDT];
    __shared__ bf16_t sWh[2][64 * LDT];
    __shared__ bf16_t sWl[2][64 * LDT];

    const int tid  = threadIdx.x;
    const int lane = tid & 31;
    const int w    = tid >> 5;
    const int half = lane >> 4;
    const int l16  = lane & 15;

    const int bm = blockIdx.y * 64;
    const int bn = blockIdx.x * 64;
    const int wm = (w & 3) * 16;   // wave row tile within block
    const int wn = (w >> 2) * 32;  // wave col strip within block

    const int ldr    = tid >> 2;       // 0..63 : tile row staged by this thread
    const int ldc    = (tid & 3) * 8;  // 0,8,16,24 : k-offset (8 bf16 = 16B)
    const int ldsOff = ldr * LDT + ldc;

    const bf16_t* gAh = Ahi + (size_t)(bm + ldr) * K + ldc;
    const bf16_t* gAl = Alo + (size_t)(bm + ldr) * K + ldc;
    const bf16_t* gWh = Whi + (size_t)(bn + ldr) * K + ldc;
    const bf16_t* gWl = Wlo + (size_t)(bn + ldr) * K + ldc;

#define STAGE(buf, koff)                                    \
    {                                                       \
        async_g2l_b128(gAh + (koff), &sAh[buf][ldsOff]);    \
        async_g2l_b128(gAl + (koff), &sAl[buf][ldsOff]);    \
        async_g2l_b128(gWh + (koff), &sWh[buf][ldsOff]);    \
        async_g2l_b128(gWl + (koff), &sWl[buf][ldsOff]);    \
    }

    v8f acc0 = {};
    v8f acc1 = {};

    STAGE(0, 0);
    const int nk = K / 32;
    for (int kc = 0; kc < nk; ++kc) {
        const int cur = kc & 1;
        __syncthreads();  // everyone done reading the buffer we refill next
        if (kc + 1 < nk) {
            STAGE(1 - cur, (kc + 1) * 32);
            WAIT_ASYNC(4);  // 4 newest outstanding -> current buffer landed
        } else {
            WAIT_ASYNC(0);
        }
        __syncthreads();  // current buffer visible to all waves

        // ---- build fragments (contiguous k = 16*half + s, same for A and B) ----
        FragBF ah, al, bh0, bl0, bh1, bl1;
        {
            const int ar = wm + l16;  // A: lane row = lane%16, K split at 16*half
            ah.q[0] = *(const uint4*)&sAh[cur][ar * LDT + 16 * half];
            ah.q[1] = *(const uint4*)&sAh[cur][ar * LDT + 16 * half + 8];
            al.q[0] = *(const uint4*)&sAl[cur][ar * LDT + 16 * half];
            al.q[1] = *(const uint4*)&sAl[cur][ar * LDT + 16 * half + 8];
            const int n0 = wn + l16;  // B: lane col = lane%16, K split at 16*half
            const int n1 = wn + 16 + l16;
            bh0.q[0] = *(const uint4*)&sWh[cur][n0 * LDT + 16 * half];
            bh0.q[1] = *(const uint4*)&sWh[cur][n0 * LDT + 16 * half + 8];
            bl0.q[0] = *(const uint4*)&sWl[cur][n0 * LDT + 16 * half];
            bl0.q[1] = *(const uint4*)&sWl[cur][n0 * LDT + 16 * half + 8];
            bh1.q[0] = *(const uint4*)&sWh[cur][n1 * LDT + 16 * half];
            bh1.q[1] = *(const uint4*)&sWh[cur][n1 * LDT + 16 * half + 8];
            bl1.q[0] = *(const uint4*)&sWl[cur][n1 * LDT + 16 * half];
            bl1.q[1] = *(const uint4*)&sWl[cur][n1 * LDT + 16 * half + 8];
        }

        // ---- bf16x3: hi*hi + hi*lo + lo*hi ----
        acc0 = wmma_bf16(ah.v, bh0.v, acc0);
        acc0 = wmma_bf16(ah.v, bl0.v, acc0);
        acc0 = wmma_bf16(al.v, bh0.v, acc0);
        acc1 = wmma_bf16(ah.v, bh1.v, acc1);
        acc1 = wmma_bf16(ah.v, bl1.v, acc1);
        acc1 = wmma_bf16(al.v, bh1.v, acc1);
    }
#undef STAGE

    // ---- epilogue: bias + store (C/D layout: M = r + 8*half, N = lane%16) ----
    const int cn0 = bn + wn + l16;
    const int cn1 = cn0 + 16;
    const float bv0 = bias[cn0];
    const float bv1 = bias[cn1];
#pragma unroll
    for (int r = 0; r < 8; ++r) {
        const int m = bm + wm + r + 8 * half;
        C[(size_t)m * N + cn0] = (OUT)(acc0[r] + bv0);
        C[(size_t)m * N + cn1] = (OUT)(acc1[r] + bv1);
    }
}

// ---------------------------------------------------------------------------
// Flash attention (causal), head_dim=64, bf16 WMMA core, fp32 online softmax.
// Grid: (S/128, B*H). 256 threads = 8 waves; wave handles 16 query rows.
// K chunk staged via async global->LDS; V staged transposed via VGPRs.
// ---------------------------------------------------------------------------
#define LDK 72  // padded LDS row stride for the 64-wide K tile

__global__ __launch_bounds__(256) void flash_attn(
    const bf16_t* __restrict__ Qb, const bf16_t* __restrict__ Kb,
    const bf16_t* __restrict__ Vb,
    bf16_t* __restrict__ Ohi, bf16_t* __restrict__ Olo) {
    const int S = 2048, DM = 1024, HD = 64;

    __shared__ bf16_t sK[32 * LDK];     // K chunk [key][d]
    __shared__ bf16_t sVt[64 * LDT];    // V chunk transposed [d][key]
    __shared__ bf16_t sP[8][16 * LDT];  // per-wave P re-layout buffer

    const int tid  = threadIdx.x;
    const int lane = tid & 31;
    const int w    = tid >> 5;
    const int half = lane >> 4;
    const int l16  = lane & 15;

    const int bh = blockIdx.y;
    const int b  = bh >> 4;
    const int h  = bh & 15;
    const size_t base = (size_t)b * S * DM + (size_t)h * HD;

    const int q0 = blockIdx.x * 128;
    const int qw = q0 + 16 * w;  // this wave's first query row
    const int qw_u = __builtin_amdgcn_readfirstlane(qw);  // scalar copy

    // Q fragments for d-chunks [0,32) and [32,64), loaded once from global.
    FragBF qf0, qf1;
    {
        const bf16_t* qrow = Qb + base + (size_t)(qw + l16) * DM;
        qf0.q[0] = *(const uint4*)(qrow + 16 * half);
        qf0.q[1] = *(const uint4*)(qrow + 16 * half + 8);
        qf1.q[0] = *(const uint4*)(qrow + 32 + 16 * half);
        qf1.q[1] = *(const uint4*)(qrow + 32 + 16 * half + 8);
    }

    v8f o0 = {}, o1 = {}, o2 = {}, o3 = {};
    float mrow[8], lrow[8];
#pragma unroll
    for (int r = 0; r < 8; ++r) { mrow[r] = -3.0e38f; lrow[r] = 0.0f; }

    const int kend = q0 + 128;       // causal bound for this block
    const int krow = tid >> 3;       // 0..31 staging row
    const int kseg = (tid & 7) * 8;  // 16B segment within 64-elem row

    for (int k0 = 0; k0 < kend; k0 += 32) {
        __syncthreads();
        // ---- cooperative stage: K async, V transposed through VGPRs ----
        {
            const size_t roff = base + (size_t)(k0 + krow) * DM + kseg;
            async_g2l_b128(&Kb[roff], &sK[krow * LDK + kseg]);
            uint4 vv = *(const uint4*)&Vb[roff];
            const bf16_t* ve = (const bf16_t*)&vv;
#pragma unroll
            for (int j = 0; j < 8; ++j)
                sVt[(kseg + j) * LDT + krow] = ve[j];
            WAIT_ASYNC(0);
        }
        __syncthreads();

        if (k0 < qw_u + 16) {  // scalar causal early-out (EXEC untouched)
            // ---- scores S = Q @ K^T for 2 key tiles of 16 ----
            FragBF kb00, kb01, kb10, kb11;
            {
                const int n0 = l16, n1 = 16 + l16;
                kb00.q[0] = *(const uint4*)&sK[n0 * LDK + 16 * half];
                kb00.q[1] = *(const uint4*)&sK[n0 * LDK + 16 * half + 8];
                kb01.q[0] = *(const uint4*)&sK[n0 * LDK + 32 + 16 * half];
                kb01.q[1] = *(const uint4*)&sK[n0 * LDK + 32 + 16 * half + 8];
                kb10.q[0] = *(const uint4*)&sK[n1 * LDK + 16 * half];
                kb10.q[1] = *(const uint4*)&sK[n1 * LDK + 16 * half + 8];
                kb11.q[0] = *(const uint4*)&sK[n1 * LDK + 32 + 16 * half];
                kb11.q[1] = *(const uint4*)&sK[n1 * LDK + 32 + 16 * half + 8];
            }
            v8f s0 = {}, s1 = {};
            s0 = wmma_bf16(qf0.v, kb00.v, s0);
            s0 = wmma_bf16(qf1.v, kb01.v, s0);
            s1 = wmma_bf16(qf0.v, kb10.v, s1);
            s1 = wmma_bf16(qf1.v, kb11.v, s1);

            // ---- online softmax (fp32); row stats via ds_swizzle butterfly ----
            float p0[8], p1[8];
#pragma unroll
            for (int r = 0; r < 8; ++r) {
                const int qr = qw + r + 8 * half;
                const int c0 = k0 + l16;
                const int c1 = k0 + 16 + l16;
                float v0 = (c0 <= qr) ? s0[r] * 0.125f : -3.0e38f;
                float v1 = (c1 <= qr) ? s1[r] * 0.125f : -3.0e38f;
                float rm = fmaxf(v0, v1);
                rm = fmaxf(rm, SWZ_F(rm, SWZ_X1));
                rm = fmaxf(rm, SWZ_F(rm, SWZ_X2));
                rm = fmaxf(rm, SWZ_F(rm, SWZ_X4));
                rm = fmaxf(rm, SWZ_F(rm, SWZ_X8));
                const float mnew  = fmaxf(mrow[r], rm);
                const float alpha = __expf(mrow[r] - mnew);
                p0[r] = __expf(v0 - mnew);
                p1[r] = __expf(v1 - mnew);
                float rs = p0[r] + p1[r];
                rs += SWZ_F(rs, SWZ_X1);
                rs += SWZ_F(rs, SWZ_X2);
                rs += SWZ_F(rs, SWZ_X4);
                rs += SWZ_F(rs, SWZ_X8);
                lrow[r] = lrow[r] * alpha + rs;
                mrow[r] = mnew;
                o0[r] *= alpha; o1[r] *= alpha; o2[r] *= alpha; o3[r] *= alpha;
            }

            // ---- re-layout P: C-frag -> LDS -> A-frag (per-wave buffer) ----
            bf16_t* pw = sP[w];
#pragma unroll
            for (int r = 0; r < 8; ++r) {
                const int row = r + 8 * half;
                pw[row * LDT + l16]      = (bf16_t)p0[r];
                pw[row * LDT + 16 + l16] = (bf16_t)p1[r];
            }
            asm volatile("s_wait_dscnt 0" ::: "memory");
            FragBF pf;
            pf.q[0] = *(const uint4*)&pw[l16 * LDT + 16 * half];
            pf.q[1] = *(const uint4*)&pw[l16 * LDT + 16 * half + 8];

            // ---- O += P @ V over 4 d-groups of 16 ----
#define FLASH_PV(g, OACC)                                                 \
            {                                                             \
                FragBF vbf;                                               \
                const int d_ = (g) * 16 + l16;                            \
                vbf.q[0] = *(const uint4*)&sVt[d_ * LDT + 16 * half];     \
                vbf.q[1] = *(const uint4*)&sVt[d_ * LDT + 16 * half + 8]; \
                OACC = wmma_bf16(pf.v, vbf.v, OACC);                      \
            }
            FLASH_PV(0, o0)
            FLASH_PV(1, o1)
            FLASH_PV(2, o2)
            FLASH_PV(3, o3)
#undef FLASH_PV
        }
    }

    // ---- finalize: O / l, write bf16 hi/lo split ----
#pragma unroll
    for (int r = 0; r < 8; ++r) {
        const float inv = 1.0f / lrow[r];
        const int qr = qw + r + 8 * half;
        const size_t rowoff = base + (size_t)qr * DM;
#define FLASH_ST(g, OACC)                                           \
        {                                                           \
            float v = OACC[r] * inv;                                \
            bf16_t hv = (bf16_t)v;                                  \
            Ohi[rowoff + (g) * 16 + l16] = hv;                      \
            Olo[rowoff + (g) * 16 + l16] = (bf16_t)(v - (float)hv); \
        }
        FLASH_ST(0, o0)
        FLASH_ST(1, o1)
        FLASH_ST(2, o2)
        FLASH_ST(3, o3)
#undef FLASH_ST
    }
}

// ---------------------------------------------------------------------------
extern "C" void kernel_launch(void* const* d_in, const int* in_sizes, int n_in,
                              void* d_out, int out_size, void* d_ws, size_t ws_size,
                              hipStream_t stream) {
    (void)in_sizes; (void)n_in; (void)out_size; (void)ws_size;
    const float* x  = (const float*)d_in[0];
    const float* wq = (const float*)d_in[1];
    const float* bq = (const float*)d_in[2];
    const float* wk = (const float*)d_in[3];
    const float* bk = (const float*)d_in[4];
    const float* wv = (const float*)d_in[5];
    const float* bv = (const float*)d_in[6];
    const float* wo = (const float*)d_in[7];
    const float* bo = (const float*)d_in[8];
    float* out = (float*)d_out;

    const int M = 4096, N = 1024, K = 1024;  // B*S x d_model
    const size_t XW = (size_t)M * K;         // activation elems
    const size_t WW = (size_t)N * K;         // weight elems

    char* ws = (char*)d_ws;
    size_t off = 0;
    auto alloc = [&](size_t bytes) {
        char* p = ws + off;
        off += (bytes + 255) & ~(size_t)255;
        return p;
    };
    bf16_t* xhi  = (bf16_t*)alloc(XW * 2);
    bf16_t* xlo  = (bf16_t*)alloc(XW * 2);
    bf16_t* wqhi = (bf16_t*)alloc(WW * 2);
    bf16_t* wqlo = (bf16_t*)alloc(WW * 2);
    bf16_t* wkhi = (bf16_t*)alloc(WW * 2);
    bf16_t* wklo = (bf16_t*)alloc(WW * 2);
    bf16_t* wvhi = (bf16_t*)alloc(WW * 2);
    bf16_t* wvlo = (bf16_t*)alloc(WW * 2);
    bf16_t* wohi = (bf16_t*)alloc(WW * 2);
    bf16_t* wolo = (bf16_t*)alloc(WW * 2);
    bf16_t* Qb   = (bf16_t*)alloc(XW * 2);
    bf16_t* Kbuf = (bf16_t*)alloc(XW * 2);
    bf16_t* Vbuf = (bf16_t*)alloc(XW * 2);
    bf16_t* ohi  = (bf16_t*)alloc(XW * 2);
    bf16_t* olo  = (bf16_t*)alloc(XW * 2);

    // 1) fp32 -> bf16 hi/lo splits
    split_f32<<<1024, 256, 0, stream>>>(x, xhi, xlo, (int)XW);
    split_f32<<<512, 256, 0, stream>>>(wq, wqhi, wqlo, (int)WW);
    split_f32<<<512, 256, 0, stream>>>(wk, wkhi, wklo, (int)WW);
    split_f32<<<512, 256, 0, stream>>>(wv, wvhi, wvlo, (int)WW);
    split_f32<<<512, 256, 0, stream>>>(wo, wohi, wolo, (int)WW);

    // 2) Q/K/V projections (bf16x3, bf16 output for the attention core)
    dim3 gg(N / 64, M / 64);
    gemm_nt_bf16x3<bf16_t><<<gg, 256, 0, stream>>>(xhi, xlo, wqhi, wqlo, bq, Qb,   M, N, K);
    gemm_nt_bf16x3<bf16_t><<<gg, 256, 0, stream>>>(xhi, xlo, wkhi, wklo, bk, Kbuf, M, N, K);
    gemm_nt_bf16x3<bf16_t><<<gg, 256, 0, stream>>>(xhi, xlo, wvhi, wvlo, bv, Vbuf, M, N, K);

    // 3) causal flash attention, writes hi/lo split directly
    dim3 fg(2048 / 128, 2 * 16);
    flash_attn<<<fg, 256, 0, stream>>>(Qb, Kbuf, Vbuf, ohi, olo);

    // 4) output projection (bf16x3, fp32 output)
    gemm_nt_bf16x3<float><<<gg, 256, 0, stream>>>(ohi, olo, wohi, wolo, bo, out, M, N, K);
}
